// Decoder_12833362281061
// MI455X (gfx1250) — matrix operand order repaired and checked by reading, not compiled, not run on hardware
//
#include <hip/hip_runtime.h>
#include <hip/hip_bf16.h>

// Problem constants (match reference)
#define BB 512
#define LL 100
#define EE 256
#define HH 256
#define TT 100
#define G4H 1024
#define C_EXPLORE 10.0f

typedef __attribute__((ext_vector_type(16))) _Float16 v16h;
typedef __attribute__((ext_vector_type(8)))  _Float16 v8h;
typedef __attribute__((ext_vector_type(8)))  float    v8f;

union H16 { v16h v; v8h h[2]; };

// ---------------------------------------------------------------------------
// WMMA tile accumulate: acc(16x16) += A[m0:m0+16, :K] * B, where
// B[k][n] = W[n0+n][k]  (i.e. computes X @ W^T with W row-major, ldw = K-ld).
// Layouts per CDNA5 ISA 7.12.2 (wave32, 16-bit A 16x32 / B 32x16).
// ---------------------------------------------------------------------------
template <int K>
__device__ __forceinline__ void wmma_accum(const _Float16* __restrict__ A, int lda, int m0,
                                           const _Float16* __restrict__ W, int ldw, int n0,
                                           v8f& acc) {
  const int lane = threadIdx.x & 31;
  const int r    = lane & 15;
  const int hi   = lane >> 4;  // 0 for lanes 0-15, 1 for lanes 16-31
  const _Float16* ap = A + (size_t)(m0 + r) * lda + hi * 8;   // A-frag base
  const _Float16* bp = W + (size_t)(n0 + r) * ldw + hi * 8;   // B-frag base
#pragma unroll
  for (int k = 0; k < K; k += 32) {
    H16 a, b;
    a.h[0] = *(const v8h*)(ap + k);        // K {0..7} / {8..15}
    a.h[1] = *(const v8h*)(ap + k + 16);   // K {16..23} / {24..31}
    b.h[0] = *(const v8h*)(bp + k);        // K {0..7} / {8..15}
    b.h[1] = *(const v8h*)(bp + k + 16);   // K {16..23} / {24..31}
    acc = __builtin_amdgcn_wmma_f32_16x16x32_f16(false, a.v, false, b.v,
                                                 (short)0, acc, false, false);
  }
}

// ---------------------------------------------------------------------------
// Elementwise helpers
// ---------------------------------------------------------------------------
__global__ void cvt_f32_f16_kernel(const float* __restrict__ src,
                                   _Float16* __restrict__ dst, int n) {
  for (int i = blockIdx.x * blockDim.x + threadIdx.x; i < n;
       i += gridDim.x * blockDim.x)
    dst[i] = (_Float16)src[i];
}

__global__ void copy_f32_kernel(const float* __restrict__ src,
                                float* __restrict__ dst, int n) {
  for (int i = blockIdx.x * blockDim.x + threadIdx.x; i < n;
       i += gridDim.x * blockDim.x)
    dst[i] = src[i];
}

// ---------------------------------------------------------------------------
// e = Wref(256x256) @ ctx^T + bref, stored transposed as e[b][l][o] (f32).
// A = Wref (M=256=o, K=256=h), B cols = context rows (n = l*B + b), N=51200.
// Tiles: 16 (M) x 3200 (N). 4 waves/block -> 12800 blocks x 128 thr. Exact.
// ---------------------------------------------------------------------------
__global__ void e_gemm_kernel(const _Float16* __restrict__ Wref,
                              const _Float16* __restrict__ ctx,
                              const float* __restrict__ bref,
                              float* __restrict__ e) {
  int wid = (blockIdx.x * blockDim.x + threadIdx.x) >> 5;
  int tm = wid & 15;        // 256/16
  int tn = wid >> 4;        // 51200/16 = 3200
  v8f acc = {};
  wmma_accum<HH>(Wref, HH, tm * 16, ctx, HH, tn * 16, acc);
  const int lane = threadIdx.x & 31;
  const int n  = tn * 16 + (lane & 15);        // pair index = l*B + b
  const int l  = n >> 9;                       // /512
  const int b  = n & 511;
  const int m0 = tm * 16 + ((lane >> 4) << 3); // output-feature row base
  float* ep = e + ((size_t)b * LL + l) * HH + m0;
#pragma unroll
  for (int j = 0; j < 8; j++) ep[j] = acc[j] + bref[m0 + j];
}

// ---------------------------------------------------------------------------
// gates(512x1024) = x@Wi^T + h@Wh^T  (biases added in LSTM kernel).
// Tiles 32x64 = 2048 waves -> 512 blocks x 128 thr. Exact.
// ---------------------------------------------------------------------------
__global__ void gates_gemm_kernel(const _Float16* __restrict__ x,
                                  const _Float16* __restrict__ h,
                                  const _Float16* __restrict__ Wi,
                                  const _Float16* __restrict__ Wh,
                                  float* __restrict__ gates) {
  int wid = (blockIdx.x * blockDim.x + threadIdx.x) >> 5;
  int tm = wid & 31;        // 512/16
  int tn = wid >> 5;        // 1024/16 = 64
  v8f acc = {};
  wmma_accum<EE>(x, EE, tm * 16, Wi, EE, tn * 16, acc);
  wmma_accum<HH>(h, HH, tm * 16, Wh, HH, tn * 16, acc);
  const int lane = threadIdx.x & 31;
  const int col  = tn * 16 + (lane & 15);
  const int row0 = tm * 16 + ((lane >> 4) << 3);
#pragma unroll
  for (int j = 0; j < 8; j++)
    gates[(size_t)(row0 + j) * G4H + col] = acc[j];
}

// ---------------------------------------------------------------------------
// qp(512x256) = X@Wq^T + bq. Tiles 32x16 = 512 waves -> 128 blocks. Exact.
// ---------------------------------------------------------------------------
__global__ void qp_gemm_kernel(const _Float16* __restrict__ X,
                               const _Float16* __restrict__ Wq,
                               const float* __restrict__ bq,
                               float* __restrict__ qp) {
  int wid = (blockIdx.x * blockDim.x + threadIdx.x) >> 5;
  int tm = wid & 31;
  int tn = wid >> 5;        // 16
  v8f acc = {};
  wmma_accum<HH>(X, HH, tm * 16, Wq, HH, tn * 16, acc);
  const int lane = threadIdx.x & 31;
  const int col  = tn * 16 + (lane & 15);
  const int row0 = tm * 16 + ((lane >> 4) << 3);
  const float bb = bq[col];
#pragma unroll
  for (int j = 0; j < 8; j++)
    qp[(size_t)(row0 + j) * HH + col] = acc[j] + bb;
}

// ---------------------------------------------------------------------------
// LSTM elementwise: one thread per (b,u). c updated in place.
// ---------------------------------------------------------------------------
__global__ void lstm_kernel(const float* __restrict__ gates,
                            const float* __restrict__ bi,
                            const float* __restrict__ bh,
                            float* __restrict__ c,
                            float* __restrict__ hy_f,
                            _Float16* __restrict__ h_h) {
  int t = blockIdx.x * blockDim.x + threadIdx.x;  // B*H
  int b = t >> 8;
  int u = t & 255;
  const float* g = gates + (size_t)b * G4H;
  float ig = g[u]         + bi[u]         + bh[u];
  float fg = g[HH + u]    + bi[HH + u]    + bh[HH + u];
  float cg = g[2*HH + u]  + bi[2*HH + u]  + bh[2*HH + u];
  float og = g[3*HH + u]  + bi[3*HH + u]  + bh[3*HH + u];
  float si = 1.f / (1.f + expf(-ig));
  float sf = 1.f / (1.f + expf(-fg));
  float so = 1.f / (1.f + expf(-og));
  float cy = sf * c[t] + si * tanhf(cg);
  float hy = so * tanhf(cy);
  c[t] = cy;
  hy_f[t] = hy;
  h_h[t] = (_Float16)hy;
}

// ---------------------------------------------------------------------------
// Glimpse attention: one block per batch row b (256 threads = 8 waves).
// logits -> mask update -> softmax -> weighted sum of e_g -> g_l.
// ---------------------------------------------------------------------------
__global__ void glimpse_kernel(const float* __restrict__ qp,
                               const float* __restrict__ e,
                               const float* __restrict__ v,
                               int* __restrict__ mask,
                               const int* __restrict__ prev,
                               int step,
                               float* __restrict__ gl_f,
                               _Float16* __restrict__ gl_h) {
  __shared__ float slg[128];
  __shared__ int   smask[128];
  __shared__ float sred[2];
  const int b    = blockIdx.x;
  const int lane = threadIdx.x & 31;
  const int wave = threadIdx.x >> 5;

  float vr[8], qr[8];
#pragma unroll
  for (int j = 0; j < 8; j++) {
    vr[j] = v[lane * 8 + j];
    qr[j] = qp[(size_t)b * HH + lane * 8 + j];
  }
  for (int l = wave; l < LL; l += 8) {
    const float* ep = e + ((size_t)b * LL + l) * HH + lane * 8;
    float s = 0.f;
#pragma unroll
    for (int j = 0; j < 8; j++) s += vr[j] * tanhf(qr[j] + ep[j]);
    for (int o = 16; o; o >>= 1) s += __shfl_xor(s, o, 32);
    if (lane == 0) slg[l] = s;
  }
  if (threadIdx.x < LL) smask[threadIdx.x] = mask[(size_t)b * LL + threadIdx.x];
  __syncthreads();
  if (threadIdx.x == 0) {          // apply_mask carry update
    if (step == 0) smask[0] = 1;
    else { if (step == 1) smask[0] = 0; smask[prev[b]] = 1; }
  }
  __syncthreads();
  if (threadIdx.x < LL) {
    mask[(size_t)b * LL + threadIdx.x] = smask[threadIdx.x];
    if (smask[threadIdx.x]) slg[threadIdx.x] = -1e30f;
  }
  __syncthreads();
  if (threadIdx.x < 32) {
    float m = -1e30f;
    for (int l = lane; l < LL; l += 32) m = fmaxf(m, slg[l]);
    for (int o = 16; o; o >>= 1) m = fmaxf(m, __shfl_xor(m, o, 32));
    if (lane == 0) sred[0] = m;
  }
  __syncthreads();
  const float mx = sred[0];
  if (threadIdx.x < LL) slg[threadIdx.x] = expf(slg[threadIdx.x] - mx);
  __syncthreads();
  if (threadIdx.x < 32) {
    float s = 0.f;
    for (int l = lane; l < LL; l += 32) s += slg[l];
    for (int o = 16; o; o >>= 1) s += __shfl_xor(s, o, 32);
    if (lane == 0) sred[1] = s;
  }
  __syncthreads();
  const float inv = 1.f / sred[1];
  const int h = threadIdx.x;                 // 256 threads = H
  float acc = 0.f;
  const float* eb = e + (size_t)b * LL * HH + h;
  for (int l = 0; l < LL; l++) acc += eb[(size_t)l * HH] * slg[l];
  acc *= inv;
  gl_f[(size_t)b * HH + h] = acc;
  gl_h[(size_t)b * HH + h] = (_Float16)acc;
}

// ---------------------------------------------------------------------------
// Pointer attention: logits -> C*tanh -> mask (read-only; idempotent after
// glimpse's update) -> softmax -> probs out, argmax -> prev/sel, gather next x.
// ---------------------------------------------------------------------------
__global__ void pointer_kernel(const float* __restrict__ qp,
                               const float* __restrict__ e,
                               const float* __restrict__ v,
                               const int* __restrict__ mask,
                               int* __restrict__ prev,
                               int step,
                               float* __restrict__ out_probs,
                               float* __restrict__ out_sel,
                               const float* __restrict__ embedded,
                               _Float16* __restrict__ x_h) {
  __shared__ float slg[128];
  __shared__ float sred[2];
  __shared__ int   sidx;
  const int b    = blockIdx.x;
  const int lane = threadIdx.x & 31;
  const int wave = threadIdx.x >> 5;

  float vr[8], qr[8];
#pragma unroll
  for (int j = 0; j < 8; j++) {
    vr[j] = v[lane * 8 + j];
    qr[j] = qp[(size_t)b * HH + lane * 8 + j];
  }
  for (int l = wave; l < LL; l += 8) {
    const float* ep = e + ((size_t)b * LL + l) * HH + lane * 8;
    float s = 0.f;
#pragma unroll
    for (int j = 0; j < 8; j++) s += vr[j] * tanhf(qr[j] + ep[j]);
    for (int o = 16; o; o >>= 1) s += __shfl_xor(s, o, 32);
    if (lane == 0) slg[l] = C_EXPLORE * tanhf(s);
  }
  __syncthreads();
  if (threadIdx.x < LL && mask[(size_t)b * LL + threadIdx.x])
    slg[threadIdx.x] = -1e30f;
  __syncthreads();
  if (threadIdx.x < 32) {
    float m = -1e30f;
    for (int l = lane; l < LL; l += 32) m = fmaxf(m, slg[l]);
    for (int o = 16; o; o >>= 1) m = fmaxf(m, __shfl_xor(m, o, 32));
    if (lane == 0) sred[0] = m;
  }
  __syncthreads();
  const float mx = sred[0];
  if (threadIdx.x < LL) slg[threadIdx.x] = expf(slg[threadIdx.x] - mx);
  __syncthreads();
  if (threadIdx.x < 32) {
    float s = 0.f;
    for (int l = lane; l < LL; l += 32) s += slg[l];
    for (int o = 16; o; o >>= 1) s += __shfl_xor(s, o, 32);
    if (lane == 0) sred[1] = s;
  }
  __syncthreads();
  const float inv = 1.f / sred[1];
  if (threadIdx.x < LL)
    out_probs[((size_t)step * BB + b) * LL + threadIdx.x] = slg[threadIdx.x] * inv;
  // argmax (first-max tie-break) over unnormalized exp values (same argmax)
  if (threadIdx.x < 32) {
    float bv = -1e38f; int bidx = 0;
    for (int l = lane; l < LL; l += 32) {
      float p = slg[l];
      if (p > bv) { bv = p; bidx = l; }
    }
    for (int o = 16; o; o >>= 1) {
      float ov = __shfl_xor(bv, o, 32);
      int   oi = __shfl_xor(bidx, o, 32);
      if (ov > bv || (ov == bv && oi < bidx)) { bv = ov; bidx = oi; }
    }
    if (lane == 0) {
      sidx = bidx;
      prev[b] = bidx;
      out_sel[(size_t)step * BB + b] = (float)bidx;
    }
  }
  __syncthreads();
  const int idx = sidx;
  const int ei = threadIdx.x;                // 256 threads = E
  x_h[(size_t)b * EE + ei] =
      (_Float16)embedded[((size_t)idx * BB + b) * EE + ei];
}

// ---------------------------------------------------------------------------
// Host launcher
// ---------------------------------------------------------------------------
extern "C" void kernel_launch(void* const* d_in, const int* in_sizes, int n_in,
                              void* d_out, int out_size, void* d_ws, size_t ws_size,
                              hipStream_t stream) {
  const float* decoder_input = (const float*)d_in[0];
  const float* embedded      = (const float*)d_in[1];
  const float* h0            = (const float*)d_in[2];
  const float* c0            = (const float*)d_in[3];
  const float* context       = (const float*)d_in[4];
  const float* Wi            = (const float*)d_in[5];
  const float* bi            = (const float*)d_in[6];
  const float* Wh            = (const float*)d_in[7];
  const float* bh            = (const float*)d_in[8];
  const float* gWq           = (const float*)d_in[9];
  const float* gbq           = (const float*)d_in[10];
  const float* gWref         = (const float*)d_in[11];
  const float* gbref         = (const float*)d_in[12];
  const float* gv            = (const float*)d_in[13];
  const float* pWq           = (const float*)d_in[14];
  const float* pbq           = (const float*)d_in[15];
  const float* pWref         = (const float*)d_in[16];
  const float* pbref         = (const float*)d_in[17];
  const float* pv            = (const float*)d_in[18];

  float* out_probs = (float*)d_out;                          // T*B*L
  float* out_sel   = out_probs + (size_t)TT * BB * LL;       // T*B
  float* out_h     = out_sel + (size_t)TT * BB;              // B*H
  float* out_c     = out_h + (size_t)BB * HH;                // B*H

  // Workspace sub-allocation (256B aligned)
  char* wsb = (char*)d_ws;
  size_t off = 0;
  auto alloc = [&](size_t bytes) -> void* {
    void* p = wsb + off;
    off += (bytes + 255) & ~(size_t)255;
    return p;
  };
  _Float16* Wi_h    = (_Float16*)alloc((size_t)G4H * EE * 2);
  _Float16* Wh_h    = (_Float16*)alloc((size_t)G4H * HH * 2);
  _Float16* gWq_h   = (_Float16*)alloc((size_t)HH * HH * 2);
  _Float16* pWq_h   = (_Float16*)alloc((size_t)HH * HH * 2);
  _Float16* gWref_h = (_Float16*)alloc((size_t)HH * HH * 2);
  _Float16* pWref_h = (_Float16*)alloc((size_t)HH * HH * 2);
  _Float16* ctx_h   = (_Float16*)alloc((size_t)LL * BB * HH * 2);
  _Float16* x_h     = (_Float16*)alloc((size_t)BB * EE * 2);
  _Float16* h_h     = (_Float16*)alloc((size_t)BB * HH * 2);
  _Float16* gl_h    = (_Float16*)alloc((size_t)BB * HH * 2);
  float* e_g   = (float*)alloc((size_t)BB * LL * HH * 4);    // 52.4 MB, L2-resident
  float* e_p   = (float*)alloc((size_t)BB * LL * HH * 4);    // 52.4 MB
  float* gates = (float*)alloc((size_t)BB * G4H * 4);
  float* c_cur = (float*)alloc((size_t)BB * HH * 4);
  float* hy_f  = (float*)alloc((size_t)BB * HH * 4);
  float* gl_f  = (float*)alloc((size_t)BB * HH * 4);
  float* qp_g  = (float*)alloc((size_t)BB * HH * 4);
  float* qp_p  = (float*)alloc((size_t)BB * HH * 4);
  int* mask = (int*)alloc((size_t)BB * LL * 4);
  int* prev = (int*)alloc((size_t)BB * 4);

  auto cvt = [&](const float* s, _Float16* d, int n) {
    int blocks = (n + 255) / 256; if (blocks > 4096) blocks = 4096;
    cvt_f32_f16_kernel<<<blocks, 256, 0, stream>>>(s, d, n);
  };

  // --- one-time (per launch) prep ---
  cvt(Wi, Wi_h, G4H * EE);
  cvt(Wh, Wh_h, G4H * HH);
  cvt(gWq, gWq_h, HH * HH);
  cvt(pWq, pWq_h, HH * HH);
  cvt(gWref, gWref_h, HH * HH);
  cvt(pWref, pWref_h, HH * HH);
  cvt(context, ctx_h, LL * BB * HH);
  cvt(decoder_input, x_h, BB * EE);
  cvt(h0, h_h, BB * HH);
  copy_f32_kernel<<<(BB * HH + 255) / 256, 256, 0, stream>>>(c0, c_cur, BB * HH);
  hipMemsetAsync(mask, 0, (size_t)BB * LL * 4, stream);
  hipMemsetAsync(prev, 0, (size_t)BB * 4, stream);

  // e_g / e_p precompute: 16 x 3200 tiles, 4 waves per 128-thread block
  e_gemm_kernel<<<(16 * 3200) / 4, 128, 0, stream>>>(gWref_h, ctx_h, gbref, e_g);
  e_gemm_kernel<<<(16 * 3200) / 4, 128, 0, stream>>>(pWref_h, ctx_h, pbref, e_p);

  // --- sequential decode, 100 steps ---
  for (int step = 0; step < TT; step++) {
    gates_gemm_kernel<<<(32 * 64) / 4, 128, 0, stream>>>(x_h, h_h, Wi_h, Wh_h, gates);
    lstm_kernel<<<(BB * HH) / 256, 256, 0, stream>>>(gates, bi, bh, c_cur, hy_f, h_h);
    qp_gemm_kernel<<<(32 * 16) / 4, 128, 0, stream>>>(h_h, gWq_h, gbq, qp_g);
    glimpse_kernel<<<BB, 256, 0, stream>>>(qp_g, e_g, gv, mask, prev, step, gl_f, gl_h);
    qp_gemm_kernel<<<(32 * 16) / 4, 128, 0, stream>>>(gl_h, pWq_h, pbq, qp_p);
    pointer_kernel<<<BB, 256, 0, stream>>>(qp_p, e_p, pv, mask, prev, step,
                                           out_probs, out_sel, embedded, x_h);
  }

  // final hx, cx
  hipMemcpyAsync(out_h, hy_f, (size_t)BB * HH * 4, hipMemcpyDeviceToDevice, stream);
  hipMemcpyAsync(out_c, c_cur, (size_t)BB * HH * 4, hipMemcpyDeviceToDevice, stream);
}
